// PK_1726576855275
// MI455X (gfx1250) — matrix operand order repaired and checked by reading, not compiled, not run on hardware
//
#include <hip/hip_runtime.h>

#define DIM      1024
#define HEADS    8
#define DIM_KEY  512
#define NUM_KEYS 1000
#define PKEYS    2
#define TOPK     16
#define MTOT     4096      // B*N = 2*2048
#define NOUT     8192      // PKEYS*HEADS*DIM_KEY
#define KPAD     1024      // NUM_KEYS padded to power of two

typedef unsigned short u16;
typedef __attribute__((ext_vector_type(8)))  __bf16 v8bf;
typedef __attribute__((ext_vector_type(16))) __bf16 v16bf;
typedef __attribute__((ext_vector_type(8)))  float  v8f;

__device__ __forceinline__ u16 f2bf(float f) {
  union { float f; unsigned u; } x; x.f = f;
  unsigned r = x.u + 0x7FFFu + ((x.u >> 16) & 1u);   // round-to-nearest-even
  return (u16)(r >> 16);
}

__device__ __forceinline__ v16bf cat8(v8bf lo, v8bf hi) {
  return __builtin_shufflevector(lo, hi, 0,1,2,3,4,5,6,7,8,9,10,11,12,13,14,15);
}

// Sorted-descending top-16 insertion; fully unrolled, static indices only
// (keeps tv/tk in VGPRs, no scratch). Ties keep the earlier-scanned candidate,
// matching jax.lax.top_k tie-breaking when scanned in ascending index order.
__device__ __forceinline__ void topk_insert(float (&tv)[TOPK], int (&tk)[TOPK],
                                            float s, int key) {
  if (s <= tv[TOPK - 1]) return;
  #pragma unroll
  for (int q = TOPK - 1; q >= 0; --q) {
    const int  qp       = (q > 0) ? (q - 1) : 0;
    const bool fromprev = (q > 0) && (s > tv[qp]);
    const float nv = fromprev ? tv[qp] : s;
    const int   nk = fromprev ? tk[qp] : key;
    if (s > tv[q]) { tv[q] = nv; tk[q] = nk; }
  }
}

__global__ __launch_bounds__(256) void cast_f32_to_bf16(const float* __restrict__ in,
                                                        u16* __restrict__ out, int n) {
  int i = blockIdx.x * 256 + threadIdx.x;
  if (i < n) out[i] = f2bf(in[i]);
}

// keys [p, k, h, d] fp32 -> kb [p, h, d, kpad] bf16 (zero padded keys >= 1000)
__global__ __launch_bounds__(256) void keys_rearrange(const float* __restrict__ keys,
                                                      u16* __restrict__ kb, int n) {
  int i = blockIdx.x * 256 + threadIdx.x;
  if (i >= n) return;
  const int key = i & (KPAD - 1);
  const int d   = (i >> 10) & (DIM_KEY - 1);
  const int ph  = i >> 19;
  const int h   = ph & 7;
  const int p   = ph >> 3;
  float v = 0.f;
  if (key < NUM_KEYS)
    v = keys[(((size_t)p * NUM_KEYS + key) * HEADS + h) * DIM_KEY + d];
  kb[i] = f2bf(v);
}

// q[4096,8192] = x[4096,1024] @ w_q[1024,8192], bf16 in / f32 acc / bf16 out.
// One wave -> 16x64 tile (4 WMMAs per K-step sharing the A fragment).
__global__ __launch_bounds__(256) void gemm1_qproj(const u16* __restrict__ xb,
                                                   const u16* __restrict__ wb,
                                                   u16* __restrict__ qb) {
  const int lane = threadIdx.x & 31;
  const int wave = threadIdx.x >> 5;
  const int tile = blockIdx.x * 8 + wave;      // 32768 tiles
  const int m0 = (tile >> 7) << 4;             // 128 n-tiles per m-row
  const int n0 = (tile & 127) << 6;
  const int lr = lane & 15;
  const int hi = lane >> 4;

  v8f c0 = {}, c1 = {}, c2 = {}, c3 = {};
  const u16* ar = xb + (size_t)(m0 + lr) * DIM + hi * 8;
  for (int k0 = 0; k0 < DIM; k0 += 32) {
    // A 16x32 bf16 fragment: lane half selects K 0-7/16-23 vs 8-15/24-31
    v16bf a = cat8(*(const v8bf*)(ar + k0), *(const v8bf*)(ar + k0 + 16));
    // B 32x16 fragment: lane = N column, lane half selects K 0-7/16-23 vs 8-15/24-31
    const u16* br = wb + (size_t)(k0 + hi * 8) * NOUT + n0 + lr;
    v16bf b0, b1, b2, b3;
    #pragma unroll
    for (int s = 0; s < 16; ++s) {
      const size_t o = (size_t)((s & 7) + (s >> 3) * 16) * NOUT;
      b0[s] = *(const __bf16*)(br + o);
      b1[s] = *(const __bf16*)(br + o + 16);
      b2[s] = *(const __bf16*)(br + o + 32);
      b3[s] = *(const __bf16*)(br + o + 48);
    }
    c0 = __builtin_amdgcn_wmma_f32_16x16x32_bf16(false, a, false, b0, (short)0, c0, false, false);
    c1 = __builtin_amdgcn_wmma_f32_16x16x32_bf16(false, a, false, b1, (short)0, c1, false, false);
    c2 = __builtin_amdgcn_wmma_f32_16x16x32_bf16(false, a, false, b2, (short)0, c2, false, false);
    c3 = __builtin_amdgcn_wmma_f32_16x16x32_bf16(false, a, false, b3, (short)0, c3, false, false);
  }
  #pragma unroll
  for (int r = 0; r < 8; ++r) {                // C layout: M = r + 8*hi, N = lr
    u16* o = qb + (size_t)(m0 + hi * 8 + r) * NOUT + n0 + lr;
    o[0]  = f2bf(c0[r]);
    o[16] = f2bf(c1[r]);
    o[32] = f2bf(c2[r]);
    o[48] = f2bf(c3[r]);
  }
}

// Fused sim + per-row top-16. One wave owns (p, h, 16-row block); the whole
// 512-d A slab (16 fragments) is register-resident across all 64 key tiles.
__global__ __launch_bounds__(256) void sim_topk(const u16* __restrict__ qb,
                                                const u16* __restrict__ kb,
                                                float* __restrict__ ts,
                                                int* __restrict__ ti) {
  __shared__ float sm[8][16 * 16];
  const int lane = threadIdx.x & 31;
  const int wave = threadIdx.x >> 5;
  const int wid  = blockIdx.x * 8 + wave;      // 4096 waves
  const int mb = wid & 255;
  const int h  = (wid >> 8) & 7;
  const int p  = wid >> 11;
  const int m0 = mb << 4;
  const int lr = lane & 15;
  const int hi = lane >> 4;
  const u16* A  = qb + (size_t)(p * HEADS + h) * DIM_KEY;
  const u16* Bb = kb + (size_t)(p * HEADS + h) * DIM_KEY * KPAD;
  float* smw = sm[wave];

  v16bf afr[16];
  {
    const u16* ap = A + (size_t)(m0 + lr) * NOUT + hi * 8;
    #pragma unroll
    for (int ks = 0; ks < 16; ++ks)
      afr[ks] = cat8(*(const v8bf*)(ap + ks * 32), *(const v8bf*)(ap + ks * 32 + 16));
  }

  float tv[TOPK]; int tk[TOPK];
  #pragma unroll
  for (int q = 0; q < TOPK; ++q) { tv[q] = -3.0e38f; tk[q] = 0; }

  for (int kt = 0; kt < KPAD / 16; ++kt) {
    const int key0 = kt * 16;
    v8f c = {};
    const u16* bp = Bb + (size_t)(hi * 8) * KPAD + key0 + lr;
    #pragma unroll
    for (int ks = 0; ks < 16; ++ks) {
      v16bf b;
      #pragma unroll
      for (int s = 0; s < 16; ++s)
        b[s] = *(const __bf16*)(bp + ((size_t)ks * 32 + (s & 7) + (s >> 3) * 16) * KPAD);
      c = __builtin_amdgcn_wmma_f32_16x16x32_bf16(false, afr[ks], false, b, (short)0, c, false, false);
    }
    #pragma unroll
    for (int r = 0; r < 8; ++r)
      smw[(hi * 8 + r) * 16 + lr] = c[r];
    __syncthreads();
    if (lane < 16) {
      #pragma unroll
      for (int j = 0; j < 16; ++j) {
        const int key = key0 + j;
        const float s = smw[lane * 16 + j];
        if (key < NUM_KEYS) topk_insert(tv, tk, s, key);
      }
    }
    __syncthreads();
  }
  if (lane < 16) {
    const int m = m0 + lane;
    const size_t base = (((size_t)p * MTOT + m) * HEADS + h) * TOPK;
    #pragma unroll
    for (int q = 0; q < TOPK; ++q) { ts[base + q] = tv[q]; ti[base + q] = tk[q]; }
  }
}

// Cartesian combine of the two top-16 lists, final top-16, index remap.
__global__ __launch_bounds__(256) void combine_topk(const float* __restrict__ ts,
                                                    const int* __restrict__ ti,
                                                    float* __restrict__ out) {
  const int t = blockIdx.x * 256 + threadIdx.x;
  if (t >= MTOT * HEADS) return;
  const int m = t >> 3;
  const int h = t & 7;
  const size_t b0 = (((size_t)0 * MTOT + m) * HEADS + h) * TOPK;
  const size_t b1 = (((size_t)1 * MTOT + m) * HEADS + h) * TOPK;
  float s0[TOPK], s1[TOPK]; int i0[TOPK], i1[TOPK];
  #pragma unroll
  for (int q = 0; q < TOPK; ++q) {
    s0[q] = ts[b0 + q]; i0[q] = ti[b0 + q];
    s1[q] = ts[b1 + q]; i1[q] = ti[b1 + q];
  }
  float tv[TOPK]; int tk[TOPK];
  #pragma unroll
  for (int q = 0; q < TOPK; ++q) { tv[q] = -3.0e38f; tk[q] = 0; }
  #pragma unroll
  for (int j = 0; j < TOPK; ++j) {
    #pragma unroll
    for (int l = 0; l < TOPK; ++l)
      topk_insert(tv, tk, s0[j] + s1[l], i0[j] + NUM_KEYS * i1[l]);
  }
  const size_t ob = ((size_t)m * HEADS + h) * TOPK;
  #pragma unroll
  for (int q = 0; q < TOPK; ++q) {
    out[ob + q] = tv[q];                                      // final_scores
    out[(size_t)MTOT * HEADS * TOPK + ob + q] = (float)tk[q]; // final_indices
  }
}

extern "C" void kernel_launch(void* const* d_in, const int* in_sizes, int n_in,
                              void* d_out, int out_size, void* d_ws, size_t ws_size,
                              hipStream_t stream) {
  const float* x    = (const float*)d_in[0];
  const float* w_q  = (const float*)d_in[1];
  const float* keys = (const float*)d_in[2];
  float* out = (float*)d_out;

  char* ws = (char*)d_ws;
  u16* xb = (u16*)ws;  ws += (size_t)MTOT * DIM * 2;                     //  8 MiB
  u16* wb = (u16*)ws;  ws += (size_t)DIM * NOUT * 2;                     // 16 MiB
  u16* kb = (u16*)ws;  ws += (size_t)PKEYS * HEADS * DIM_KEY * KPAD * 2; // 16 MiB
  u16* qb = (u16*)ws;  ws += (size_t)MTOT * NOUT * 2;                    // 64 MiB
  float* tsv = (float*)ws; ws += (size_t)PKEYS * MTOT * HEADS * TOPK * 4;
  int*   tiv = (int*)ws;

  {
    const int n = MTOT * DIM;
    cast_f32_to_bf16<<<(n + 255) / 256, 256, 0, stream>>>(x, xb, n);
  }
  {
    const int n = DIM * NOUT;
    cast_f32_to_bf16<<<(n + 255) / 256, 256, 0, stream>>>(w_q, wb, n);
  }
  {
    const int n = PKEYS * HEADS * DIM_KEY * KPAD;
    keys_rearrange<<<(n + 255) / 256, 256, 0, stream>>>(keys, kb, n);
  }
  gemm1_qproj<<<(MTOT / 16) * (NOUT / 64) / 8, 256, 0, stream>>>(xb, wb, qb);
  sim_topk<<<(PKEYS * HEADS * (MTOT / 16)) / 8, 256, 0, stream>>>(qb, kb, tsv, tiv);
  combine_topk<<<(MTOT * HEADS) / 256, 256, 0, stream>>>(tsv, tiv, out);
}
